// MambaLayer_24180665876798
// MI455X (gfx1250) — matrix operand order repaired and checked by reading, not compiled, not run on hardware
//
#include <hip/hip_runtime.h>
#include <cstdint>
#include <cstddef>

// ---------------- problem constants ----------------
#define B_SZ    4
#define L_SZ    2048
#define DIM     1024
#define D_INNER 2048
#define D_STATE 16
#define D_CONV  4
#define DT_RANK 64
#define M_ROWS  (B_SZ * L_SZ)            // 8192
#define XDB_N   (DT_RANK + 2 * D_STATE)  // 96
#define XDB_NP  128                      // padded to BN multiple

// ---------------- WMMA types ----------------
typedef __attribute__((ext_vector_type(16))) __bf16 v16bf;
typedef __attribute__((ext_vector_type(8)))  float  v8f;

union FragBF { v16bf v; uint4 u[2]; };
union AccF   { v8f   v; float f[8]; };

__device__ __forceinline__ unsigned short f32_to_bf16(float f) {
    unsigned int u = __float_as_uint(f);
    u += 0x7FFFu + ((u >> 16) & 1u);           // round-to-nearest-even
    return (unsigned short)(u >> 16);
}
__device__ __forceinline__ float silu_f(float x) { return x / (1.f + __expf(-x)); }

// ---- CDNA5 async global->LDS copy (ASYNCcnt) ----
__device__ __forceinline__ void async_b128(unsigned lds_off, const unsigned short* g) {
    asm volatile("global_load_async_to_lds_b128 %0, %1, off"
                 :: "v"(lds_off), "v"(g) : "memory");
}
__device__ __forceinline__ void wait_async_le4() {
    asm volatile("s_wait_asynccnt 0x4" ::: "memory");
}
__device__ __forceinline__ void wait_async_le0() {
    asm volatile("s_wait_asynccnt 0x0" ::: "memory");
}

// ---------------- fp32 -> bf16 convert ----------------
__global__ void f32_to_bf16_kernel(const float* __restrict__ in,
                                   unsigned short* __restrict__ out, int n) {
    int i = blockIdx.x * blockDim.x + threadIdx.x;
    if (i < n) out[i] = f32_to_bf16(in[i]);
}

// ---------------- fp32 [K,N] -> bf16 transposed [Npad,K] (zero pad) ----------------
__global__ void transpose_to_bf16_kernel(const float* __restrict__ in,
                                         unsigned short* __restrict__ out,
                                         int K, int N, int Npad) {
    int i = blockIdx.x * blockDim.x + threadIdx.x;   // over Npad*K
    if (i >= Npad * K) return;
    int n = i / K, k = i - n * K;
    out[i] = (n < N) ? f32_to_bf16(in[(size_t)k * N + n]) : (unsigned short)0;
}

// ---------------- generic bf16 WMMA GEMM (async double-buffered) ----------------
// C[M,N] = A[M,K](bf16, lda) @ Bt[Npad,K](bf16, ldb=K, pre-transposed) + bias[N]
// mode 0: plain (optional bf16 mirror), mode 1: softplus epilogue
#define BM 128
#define BN 128
#define BK 32
#define LDSS 48   // ushort row stride: 96B, 16B-aligned sub-chunks, spreads banks

__global__ __launch_bounds__(256)
void wmma_gemm_bf16(const unsigned short* __restrict__ A, int lda,
                    const unsigned short* __restrict__ Bt, int ldb,
                    const float* __restrict__ bias,
                    float* __restrict__ Cf, unsigned short* __restrict__ Cb,
                    int M, int N, int K, int mode)
{
    __shared__ __align__(16) unsigned short lA[2][BM][LDSS];
    __shared__ __align__(16) unsigned short lB[2][BN][LDSS];

    const int tid  = threadIdx.x;
    const int wave = tid >> 5;
    const int lane = tid & 31;
    const int half = lane >> 4;     // 0: lanes 0-15, 1: lanes 16-31
    const int l16  = lane & 15;
    const int blockM = blockIdx.x * BM;
    const int blockN = blockIdx.y * BN;

    AccF acc[8];
#pragma unroll
    for (int i = 0; i < 8; ++i) acc[i].v = (v8f){0.f,0.f,0.f,0.f,0.f,0.f,0.f,0.f};

    // tile loader mapping: each thread owns 32B (16 ushorts) of one row, A and B alike
    const int ld_row = tid >> 1;            // 0..127
    const int ld_col = (tid & 1) * 16;      // 0 / 16 (ushorts)

    auto issue_tile = [&](int k0, int bufi) {
        const unsigned short* ga = A  + (size_t)(blockM + ld_row) * lda + k0 + ld_col;
        const unsigned short* gb = Bt + (size_t)(blockN + ld_row) * ldb + k0 + ld_col;
        async_b128((unsigned)(uintptr_t)&lA[bufi][ld_row][ld_col],     ga);
        async_b128((unsigned)(uintptr_t)&lA[bufi][ld_row][ld_col + 8], ga + 8);
        async_b128((unsigned)(uintptr_t)&lB[bufi][ld_row][ld_col],     gb);
        async_b128((unsigned)(uintptr_t)&lB[bufi][ld_row][ld_col + 8], gb + 8);
    };

    const int nk = K / BK;
    issue_tile(0, 0);

    for (int kt = 0; kt < nk; ++kt) {
        const int cur = kt & 1;
        __syncthreads();                       // all waves done with buf[cur^1]
        const bool more = (kt + 1 < nk);
        if (more) issue_tile((kt + 1) * BK, cur ^ 1);
        if (more) wait_async_le4();            // in-order: current tile's 4 are done
        else      wait_async_le0();
        __syncthreads();                       // publish everyone's tile

        // A frag: lanes 0-15 K=[0..7]+[16..23]; lanes 16-31 K=[8..15]+[24..31]
        FragBF a;
        a.u[0] = *(const uint4*)&lA[cur][wave * 16 + l16][half * 8];
        a.u[1] = *(const uint4*)&lA[cur][wave * 16 + l16][half * 8 + 16];

        // preload all 8 B frags, then burst the WMMAs
        FragBF bf[8];
#pragma unroll
        for (int nt = 0; nt < 8; ++nt) {
            bf[nt].u[0] = *(const uint4*)&lB[cur][nt * 16 + l16][half * 8];
            bf[nt].u[1] = *(const uint4*)&lB[cur][nt * 16 + l16][half * 8 + 16];
        }
#pragma unroll
        for (int nt = 0; nt < 8; ++nt)
            acc[nt].v = __builtin_amdgcn_wmma_f32_16x16x32_bf16(
                false, a.v, false, bf[nt].v, (short)0, acc[nt].v, false, false);
    }

    // ---- epilogue ----
#pragma unroll
    for (int nt = 0; nt < 8; ++nt) {
        const int n = blockN + nt * 16 + l16;
        if (n >= N) continue;
        const float bv = bias ? bias[n] : 0.f;
#pragma unroll
        for (int r = 0; r < 8; ++r) {
            const int m = blockM + wave * 16 + half * 8 + r;
            float v = acc[nt].f[r] + bv;
            if (mode == 1) v = (v > 20.f) ? v : log1pf(__expf(v));   // softplus
            const size_t idx = (size_t)m * N + n;
            if (Cf) Cf[idx] = v;
            if (Cb) Cb[idx] = f32_to_bf16(v);
        }
    }
}

// ---------------- causal depthwise conv1d + SiLU ----------------
__global__ void conv_silu_kernel(const float* __restrict__ xz,
                                 const float* __restrict__ conv_w,
                                 const float* __restrict__ conv_b,
                                 float* __restrict__ xc_f32,
                                 unsigned short* __restrict__ xc_b16)
{
    const int i = blockIdx.x * blockDim.x + threadIdx.x;   // over B*L*D_INNER
    const int d  = i & (D_INNER - 1);
    const int bl = i >> 11;                                 // /D_INNER
    const int l  = bl & (L_SZ - 1);
    const int b  = bl >> 11;                                // /L_SZ
    float acc = conv_b[d];
#pragma unroll
    for (int k = 0; k < D_CONV; ++k) {
        const int ls = l - (D_CONV - 1) + k;
        if (ls >= 0)
            acc += conv_w[d * D_CONV + k] *
                   xz[((size_t)(b * L_SZ + ls)) * (2 * D_INNER) + d];  // x half of xz
    }
    const float s = silu_f(acc);
    xc_f32[i] = s;
    xc_b16[i] = f32_to_bf16(s);
}

// ---------------- selective scan (fused +D*x, *silu(z)) ----------------
__global__ __launch_bounds__(256)
void scan_kernel(const float* __restrict__ xc,
                 const float* __restrict__ dtf,
                 const float* __restrict__ xdb,
                 const float* __restrict__ xz,
                 const float* __restrict__ A_log,
                 const float* __restrict__ Dvec,
                 unsigned short* __restrict__ y_b16)
{
    const int b = blockIdx.x >> 3;                 // D_INNER/256 = 8 blocks per batch
    const int d = (blockIdx.x & 7) * 256 + threadIdx.x;

    float Arow[D_STATE];
#pragma unroll
    for (int n = 0; n < D_STATE; ++n) Arow[n] = -__expf(A_log[d * D_STATE + n]);
    const float Dd = Dvec[d];

    float s[D_STATE];
#pragma unroll
    for (int n = 0; n < D_STATE; ++n) s[n] = 0.f;

    const size_t base = (size_t)b * L_SZ;
    for (int l = 0; l < L_SZ; ++l) {
        const size_t idx = base + l;
        const float x  = xc [idx * D_INNER + d];
        const float dt = dtf[idx * D_INNER + d];
        const float* __restrict__ bc = xdb + idx * XDB_N;   // block-uniform
        const float xdt = x * dt;
        float y = 0.f;
#pragma unroll
        for (int n = 0; n < D_STATE; ++n) {
            const float dA = __expf(dt * Arow[n]);
            s[n] = s[n] * dA + xdt * bc[DT_RANK + n];            // B_t
            y += s[n] * bc[DT_RANK + D_STATE + n];               // C_t
        }
        const float z = xz[idx * (2 * D_INNER) + D_INNER + d];   // gate half
        y_b16[idx * D_INNER + d] = f32_to_bf16((y + Dd * x) * silu_f(z));
    }
}

// ---------------- launcher ----------------
extern "C" void kernel_launch(void* const* d_in, const int* in_sizes, int n_in,
                              void* d_out, int out_size, void* d_ws, size_t ws_size,
                              hipStream_t stream)
{
    const float* hidden = (const float*)d_in[0];
    const float* W_in   = (const float*)d_in[1];
    const float* b_in   = (const float*)d_in[2];
    const float* conv_w = (const float*)d_in[3];
    const float* conv_b = (const float*)d_in[4];
    const float* W_x    = (const float*)d_in[5];
    const float* W_dt   = (const float*)d_in[6];
    const float* b_dt   = (const float*)d_in[7];
    const float* A_log  = (const float*)d_in[8];
    const float* Dvec   = (const float*)d_in[9];
    const float* W_out  = (const float*)d_in[10];
    const float* b_out  = (const float*)d_in[11];
    float* out = (float*)d_out;

    // workspace carve-up (all sizes 256B-multiples)
    char* p = (char*)d_ws;
    auto grab = [&](size_t bytes) { char* q = p; p += bytes; return q; };
    unsigned short* h_b16    = (unsigned short*)grab((size_t)M_ROWS * DIM * 2);
    unsigned short* WinT     = (unsigned short*)grab((size_t)(2 * D_INNER) * DIM * 2);  // [4096,1024]
    unsigned short* WxT      = (unsigned short*)grab((size_t)XDB_NP * D_INNER * 2);     // [128,2048]
    unsigned short* WdtT     = (unsigned short*)grab((size_t)D_INNER * DT_RANK * 2);    // [2048,64]
    unsigned short* WoutT    = (unsigned short*)grab((size_t)DIM * D_INNER * 2);        // [1024,2048]
    float*          xz       = (float*)grab((size_t)M_ROWS * 2 * D_INNER * 4);
    float*          xc_f32   = (float*)grab((size_t)M_ROWS * D_INNER * 4);
    unsigned short* xc_b16   = (unsigned short*)grab((size_t)M_ROWS * D_INNER * 2);
    float*          xdb_f32  = (float*)grab((size_t)M_ROWS * XDB_N * 4);
    unsigned short* xdb_b16  = (unsigned short*)grab((size_t)M_ROWS * XDB_N * 2);
    float*          dt_f32   = (float*)grab((size_t)M_ROWS * D_INNER * 4);
    unsigned short* y_b16    = (unsigned short*)grab((size_t)M_ROWS * D_INNER * 2);

    // activation convert (row-major A side)
    {
        int n = M_ROWS * DIM;
        f32_to_bf16_kernel<<<(n + 255) / 256, 256, 0, stream>>>(hidden, h_b16, n);
    }
    // weight convert + transpose + pad (B side, [Npad,K])
    auto cvtT = [&](const float* src, unsigned short* dst, int K, int N, int Npad) {
        int n = Npad * K;
        transpose_to_bf16_kernel<<<(n + 255) / 256, 256, 0, stream>>>(src, dst, K, N, Npad);
    };
    cvtT(W_in,  WinT,  DIM,     2 * D_INNER, 2 * D_INNER);
    cvtT(W_x,   WxT,   D_INNER, XDB_N,       XDB_NP);
    cvtT(W_dt,  WdtT,  DT_RANK, D_INNER,     D_INNER);
    cvtT(W_out, WoutT, D_INNER, DIM,         DIM);

    // G1: xz = hidden @ W_in + b_in          [8192,4096] k=1024
    wmma_gemm_bf16<<<dim3(M_ROWS / BM, (2 * D_INNER) / BN), 256, 0, stream>>>(
        h_b16, DIM, WinT, DIM, b_in,
        xz, nullptr, M_ROWS, 2 * D_INNER, DIM, 0);

    // conv1d + SiLU
    conv_silu_kernel<<<(M_ROWS * D_INNER) / 256, 256, 0, stream>>>(
        xz, conv_w, conv_b, xc_f32, xc_b16);

    // G2: x_db = x_conv @ W_x                [8192,96] k=2048 (Npad=128)
    wmma_gemm_bf16<<<dim3(M_ROWS / BM, XDB_NP / BN), 256, 0, stream>>>(
        xc_b16, D_INNER, WxT, D_INNER, nullptr,
        xdb_f32, xdb_b16, M_ROWS, XDB_N, D_INNER, 0);

    // G3: dt_full = softplus(dt @ W_dt + b_dt) [8192,2048] k=64 (lda=96)
    wmma_gemm_bf16<<<dim3(M_ROWS / BM, D_INNER / BN), 256, 0, stream>>>(
        xdb_b16, XDB_N, WdtT, DT_RANK, b_dt,
        dt_f32, nullptr, M_ROWS, D_INNER, DT_RANK, 1);

    // selective scan + gating -> y (bf16)
    scan_kernel<<<B_SZ * (D_INNER / 256), 256, 0, stream>>>(
        xc_f32, dt_f32, xdb_f32, xz, A_log, Dvec, y_b16);

    // G4: out = y @ W_out + b_out            [8192,1024] k=2048
    wmma_gemm_bf16<<<dim3(M_ROWS / BM, DIM / BN), 256, 0, stream>>>(
        y_b16, D_INNER, WoutT, DIM, b_out,
        out, nullptr, M_ROWS, DIM, D_INNER, 0);
}
